// EdgeModel_48636209660177
// MI455X (gfx1250) — compile-verified, matrix operand order checked by reading, not run on hardware
//
#include <hip/hip_runtime.h>

typedef __attribute__((ext_vector_type(16))) _Float16 v16h;
typedef __attribute__((ext_vector_type(8)))  _Float16 v8h;
typedef __attribute__((ext_vector_type(8)))  float    v8f;
typedef __attribute__((ext_vector_type(4)))  float    v4f;

#define LN_EPS 1e-5f

__device__ __forceinline__ v8f wmma16x16x32(v16h a, v16h b, v8f c) {
  // D = A(16x32 f16) x B(32x16 f16) + C(16x16 f32)
  return __builtin_amdgcn_wmma_f32_16x16x32_f16(false, a, false, b, (short)0, c,
                                                false, false);
}

// bias + ReLU + LayerNorm on a 16x64 tile held in C-layout:
// lane L holds columns n = j*16 + (L%16), rows m = i + 8*(L/16) in acc[j][i].
// Row stats need a 16-lane xor-reduction (stays within each half-wave).
__device__ __forceinline__ void bias_relu_ln(v8f* acc, const float* bv,
                                             const float* gv, const float* ev) {
#pragma unroll
  for (int j = 0; j < 4; ++j)
#pragma unroll
    for (int i = 0; i < 8; ++i) {
      float v = acc[j][i] + bv[j];
      acc[j][i] = v > 0.f ? v : 0.f;
    }
#pragma unroll
  for (int i = 0; i < 8; ++i) {
    float s = acc[0][i] + acc[1][i] + acc[2][i] + acc[3][i];
    float q = acc[0][i] * acc[0][i] + acc[1][i] * acc[1][i] +
              acc[2][i] * acc[2][i] + acc[3][i] * acc[3][i];
#pragma unroll
    for (int d = 1; d < 16; d <<= 1) {
      s += __shfl_xor(s, d);
      q += __shfl_xor(q, d);
    }
    float m   = s * 0.015625f;           // /64
    float var = q * 0.015625f - m * m;
    float rs  = rsqrtf(var + LN_EPS);
#pragma unroll
    for (int j = 0; j < 4; ++j)
      acc[j][i] = (acc[j][i] - m) * rs * gv[j] + ev[j];
  }
}

// C-layout f32 tile -> f16 via wave-private LDS -> two A-layout v16h frags.
// hx is this wave's 16x72-half scratch (stride 72 halves avoids bank conflicts).
__device__ __forceinline__ void transpose_to_frags(const v8f* acc, _Float16* hx,
                                                   int lm, int hi, v16h* H) {
#pragma unroll
  for (int j = 0; j < 4; ++j)
#pragma unroll
    for (int i = 0; i < 8; ++i)
      hx[(i + 8 * hi) * 72 + j * 16 + lm] = (_Float16)acc[j][i];
  asm volatile("s_wait_dscnt 0" ::: "memory");
#pragma unroll
  for (int c = 0; c < 2; ++c) {
    const _Float16* p = hx + lm * 72 + c * 32 + hi * 8;
    v8h lo = *(const v8h*)(p);
    v8h hh = *(const v8h*)(p + 16);
    H[c] = __builtin_shufflevector(lo, hh, 0, 1, 2, 3, 4, 5, 6, 7, 8, 9, 10, 11,
                                   12, 13, 14, 15);
  }
}

__global__ __launch_bounds__(256) void edge_mlp_kernel(
    const float* __restrict__ src, const float* __restrict__ dst,
    const float* __restrict__ ea, const float* __restrict__ W1,
    const float* __restrict__ b1, const float* __restrict__ g1,
    const float* __restrict__ be1, const float* __restrict__ W2,
    const float* __restrict__ b2, const float* __restrict__ g2,
    const float* __restrict__ be2, const float* __restrict__ W3,
    const float* __restrict__ b3, float* __restrict__ out, int E, int ntiles) {
  // Weight B-fragments, packed once per block:
  //  W1: frags [0,12)  = c(0..2)*4 + j(0..3)   (96x64)
  //  W2: frags [12,32) = 12 + c(0..4)*4 + j    (160x64; c0-1 = h part, c2-4 = x0)
  //  W3: frags [32,36) = 32 + c(0..1)*2 + j(0..1) (64x32)
  __shared__ __align__(32) _Float16 wfrag[36 * 512];
  __shared__ __align__(16) _Float16 hx16[8][16 * 72];

  const int tid = threadIdx.x;

  for (int t = tid; t < 1152; t += 256) {
    const float* W;
    int ncols, c, j, fi, s;
    if (t < 384) {
      int f = t >> 5; s = t & 31; c = f >> 2; j = f & 3;
      W = W1; ncols = 64; fi = f;
    } else if (t < 1024) {
      int u = t - 384; int f = u >> 5; s = u & 31; c = f >> 2; j = f & 3;
      W = W2; ncols = 64; fi = 12 + f;
    } else {
      int u = t - 1024; int f = u >> 5; s = u & 31; c = f >> 1; j = f & 1;
      W = W3; ncols = 32; fi = 32 + f;
    }
    const int n  = j * 16 + (s & 15);
    const int h2 = s >> 4;
    _Float16* d = &wfrag[fi * 512 + s * 16];
#pragma unroll
    for (int e = 0; e < 8; ++e) {
      int k0 = c * 32 + h2 * 8 + e;   // B frag element layout matches A's K runs
      int k1 = k0 + 16;
      d[e]     = (_Float16)W[k0 * ncols + n];
      d[e + 8] = (_Float16)W[k1 * ncols + n];
    }
  }
  __syncthreads();

  const int wid  = tid >> 5;
  const int lane = tid & 31;
  const int lm   = lane & 15;
  const int hi   = lane >> 4;
  _Float16* hx = &hx16[wid][0];

  float b1v[4], g1v[4], e1v[4], b2v[4], g2v[4], e2v[4], b3v[2];
#pragma unroll
  for (int j = 0; j < 4; ++j) {
    int n = j * 16 + lm;
    b1v[j] = b1[n]; g1v[j] = g1[n]; e1v[j] = be1[n];
    b2v[j] = b2[n]; g2v[j] = g2[n]; e2v[j] = be2[n];
  }
  b3v[0] = b3[lm];
  b3v[1] = b3[16 + lm];

  const long wstride = (long)gridDim.x * 8;
  for (long tile = (long)blockIdx.x * 8 + wid; tile < ntiles; tile += wstride) {
    const long row0 = tile * 16;
    long r = row0 + lm;
    if (r > (long)E - 1) r = (long)E - 1;

    // ---- build x0 A-fragments (3 K-chunks of 32 == the 3 input arrays) ----
    v16h A0[3];
    {
      const float* bases[3] = {src, dst, ea};
#pragma unroll
      for (int c = 0; c < 3; ++c) {
        const float* p = bases[c] + r * 32 + hi * 8;
        v4f f0 = *(const v4f*)(p);
        v4f f1 = *(const v4f*)(p + 4);
        v4f f2 = *(const v4f*)(p + 16);
        v4f f3 = *(const v4f*)(p + 20);
        v16h a;
#pragma unroll
        for (int e = 0; e < 4; ++e) {
          a[e]      = (_Float16)f0[e];
          a[e + 4]  = (_Float16)f1[e];
          a[e + 8]  = (_Float16)f2[e];
          a[e + 12] = (_Float16)f3[e];
        }
        A0[c] = a;
      }
    }

    // prefetch next tile's rows (global_prefetch_b8 pulls whole cachelines)
    {
      long nt = tile + wstride;
      if (nt < ntiles) {
        long pr = nt * 16 + lm;
        __builtin_prefetch(src + pr * 32, 0, 0);
        __builtin_prefetch(dst + pr * 32, 0, 0);
        __builtin_prefetch(ea + pr * 32, 0, 0);
      }
    }

    // ---- layer 1: x0[16x96] @ W1[96x64] ----
    v8f acc[4];
#pragma unroll
    for (int j = 0; j < 4; ++j) acc[j] = (v8f){0, 0, 0, 0, 0, 0, 0, 0};
#pragma unroll
    for (int j = 0; j < 4; ++j)
#pragma unroll
      for (int c = 0; c < 3; ++c) {
        v16h b = *(const v16h*)(&wfrag[(c * 4 + j) * 512 + lane * 16]);
        acc[j] = wmma16x16x32(A0[c], b, acc[j]);
      }
    bias_relu_ln(acc, b1v, g1v, e1v);
    v16h H1[2];
    transpose_to_frags(acc, hx, lm, hi, H1);

    // ---- layer 2: cat(h1, x0)[16x160] @ W2[160x64] ----
    v8f acc2[4];
#pragma unroll
    for (int j = 0; j < 4; ++j) acc2[j] = (v8f){0, 0, 0, 0, 0, 0, 0, 0};
#pragma unroll
    for (int j = 0; j < 4; ++j) {
#pragma unroll
      for (int c = 0; c < 2; ++c) {  // h1 part: W2 rows [0,64)
        v16h b = *(const v16h*)(&wfrag[(12 + c * 4 + j) * 512 + lane * 16]);
        acc2[j] = wmma16x16x32(H1[c], b, acc2[j]);
      }
#pragma unroll
      for (int c = 0; c < 3; ++c) {  // x0 part: W2 rows [64,160)
        v16h b = *(const v16h*)(&wfrag[(12 + (2 + c) * 4 + j) * 512 + lane * 16]);
        acc2[j] = wmma16x16x32(A0[c], b, acc2[j]);
      }
    }
    bias_relu_ln(acc2, b2v, g2v, e2v);
    v16h H2[2];
    transpose_to_frags(acc2, hx, lm, hi, H2);

    // ---- layer 3: h2[16x64] @ W3[64x32] + b3 ----
    v8f acc3[2];
    acc3[0] = (v8f){0, 0, 0, 0, 0, 0, 0, 0};
    acc3[1] = (v8f){0, 0, 0, 0, 0, 0, 0, 0};
#pragma unroll
    for (int j = 0; j < 2; ++j)
#pragma unroll
      for (int c = 0; c < 2; ++c) {
        v16h b = *(const v16h*)(&wfrag[(32 + c * 2 + j) * 512 + lane * 16]);
        acc3[j] = wmma16x16x32(H2[c], b, acc3[j]);
      }
#pragma unroll
    for (int i = 0; i < 8; ++i) {
      long rr = row0 + i + 8 * hi;
      if (rr < E) {
        out[rr * 32 + lm]      = acc3[0][i] + b3v[0];
        out[rr * 32 + 16 + lm] = acc3[1][i] + b3v[1];
      }
    }
  }
}

extern "C" void kernel_launch(void* const* d_in, const int* in_sizes, int n_in,
                              void* d_out, int out_size, void* d_ws,
                              size_t ws_size, hipStream_t stream) {
  const float* src = (const float*)d_in[0];
  const float* dst = (const float*)d_in[1];
  const float* ea  = (const float*)d_in[2];
  const float* W1  = (const float*)d_in[3];
  const float* b1  = (const float*)d_in[4];
  const float* g1  = (const float*)d_in[5];
  const float* be1 = (const float*)d_in[6];
  const float* W2  = (const float*)d_in[7];
  const float* b2  = (const float*)d_in[8];
  const float* g2  = (const float*)d_in[9];
  const float* be2 = (const float*)d_in[10];
  const float* W3  = (const float*)d_in[11];
  const float* b3  = (const float*)d_in[12];
  // d_in[13] = eb (unused by reference math)
  float* out = (float*)d_out;

  const int E      = in_sizes[0] / 32;
  const int ntiles = (E + 15) / 16;
  int blocks = (ntiles + 7) / 8;
  if (blocks > 768) blocks = 768;
  if (blocks < 1) blocks = 1;

  edge_mlp_kernel<<<blocks, 256, 0, stream>>>(src, dst, ea, W1, b1, g1, be1, W2,
                                              b2, g2, be2, W3, b3, out, E,
                                              ntiles);
}
